// SelfAttention_30863634989094
// MI455X (gfx1250) — matrix operand order repaired and checked by reading, not compiled, not run on hardware
//
#include <hip/hip_runtime.h>

// ---------------------------------------------------------------------------
// CDNA5 / gfx1250 self-attention forward.
// All GEMMs use v_wmma_f32_16x16x32_bf16 (bf16 in, f32 accumulate), wave32.
// Fragment layouts used here (A and B share the same contiguous k order):
//   A 16x32: lane (g=l/16, ln=l%16) -> row M=ln, K contiguous 16g..16g+15
//   B 32x16: lane -> col N=ln, K contiguous 16g..16g+15
//   C 16x16: vgpr i, lane -> (M = i + 8g, N = ln)
// GEMMs are tiled 32x64 per wave: 2 A-frags + 4 B-frags feed 8 wmmas per
// K-step (high wmma density, ~21 FLOP/byte from L2-resident operands).
// The 64-wide N tile == one full head, so RoPE's d <-> d+-32 pairing is
// register-local (acc[s] <-> acc[s+2]): no LDS, no barriers in the epilogue.
// ---------------------------------------------------------------------------

typedef __bf16 bf16x8  __attribute__((ext_vector_type(8)));
typedef __bf16 bf16x16 __attribute__((ext_vector_type(16)));
typedef float  f32x8   __attribute__((ext_vector_type(8)));

#define Bb   2
#define Tt   2048
#define Cc   1024
#define Hh   16
#define HD   64
#define NEG_BIG_F (-1.0e10f)
#define SCALE_F   (0.125f)   // 1/sqrt(64)

__device__ __forceinline__ f32x8 wmma_bf16(bf16x16 a, bf16x16 b, f32x8 c) {
  // D = A(16x32) * B(32x16) + C, f32 accumulate
  return __builtin_amdgcn_wmma_f32_16x16x32_bf16(
      /*neg_a=*/false, a, /*neg_b=*/false, b,
      /*c_mod=*/(short)0, c, /*reuse_a=*/false, /*reuse_b=*/false);
}

__device__ __forceinline__ bf16x16 ld_afrag(const __bf16* row, int c0) {
  bf16x8 lo = *(const bf16x8*)(row + 2 * c0);        // 16B aligned
  bf16x8 hi = *(const bf16x8*)(row + 2 * c0 + 8);
  return __builtin_shufflevector(lo, hi, 0,1,2,3,4,5,6,7,8,9,10,11,12,13,14,15);
}

// ---------------------------------------------------------------------------
// Kernel 0: f32 -> bf16 conversion
// ---------------------------------------------------------------------------
__global__ void cvt_bf16_kernel(const float* __restrict__ in,
                                __bf16* __restrict__ out, int n) {
  int i = blockIdx.x * blockDim.x + threadIdx.x;
  if (i < n) out[i] = (__bf16)in[i];
}

// ---------------------------------------------------------------------------
// Kernel 1: fused QKV GEMM + RoPE, 32x64 tile per wave (one full head in N).
//   qkv[m,n] = sum_k x[m,k] * W_attn[n,k],  m in [0,4096), n in [0,3072)
//   waves = (4096/32) * (3072/64) = 6144; block = 256 (8 independent waves)
//   q,k stored (bh, t, d) bf16; v stored transposed (bh, d, t) bf16.
// ---------------------------------------------------------------------------
__global__ __launch_bounds__(256)
void qkv_rope_kernel(const __bf16* __restrict__ xb,
                     const __bf16* __restrict__ wab,
                     const float*  __restrict__ freqs,
                     __bf16* __restrict__ qout,
                     __bf16* __restrict__ kout,
                     __bf16* __restrict__ vtout) {
  const int lane = threadIdx.x & 31;
  const int wv   = threadIdx.x >> 5;
  const int g    = lane >> 4;       // lane group 0/1
  const int ln   = lane & 15;

  const int gw   = blockIdx.x * 8 + wv;   // 0..6143
  const int nblk = gw % 48;               // 64-wide N block (one head of q|k|v)
  const int mblk = gw / 48;               // 32-row M block
  const int m0   = mblk * 32;
  const int n0   = nblk * 64;

  const __bf16* arow0 = xb + (size_t)(m0 + ln) * Cc;       // rows m0..m0+15
  const __bf16* arow1 = xb + (size_t)(m0 + 16 + ln) * Cc;  // rows m0+16..m0+31
  const __bf16* brow0 = wab + (size_t)(n0 +  0 + ln) * Cc;
  const __bf16* brow1 = wab + (size_t)(n0 + 16 + ln) * Cc;
  const __bf16* brow2 = wab + (size_t)(n0 + 32 + ln) * Cc;
  const __bf16* brow3 = wab + (size_t)(n0 + 48 + ln) * Cc;
  const int c0 = g * 8;

  f32x8 acc[2][4];
#pragma unroll
  for (int ma = 0; ma < 2; ++ma)
#pragma unroll
    for (int s = 0; s < 4; ++s) acc[ma][s] = (f32x8){};

  for (int k = 0; k < Cc; k += 32) {
    __builtin_prefetch(arow0 + k + 128, 0, 3);
    __builtin_prefetch(brow0 + k + 128, 0, 3);
    bf16x16 a0 = ld_afrag(arow0 + k, c0);
    bf16x16 a1 = ld_afrag(arow1 + k, c0);
    bf16x16 b0 = *(const bf16x16*)(brow0 + k + g * 16);
    bf16x16 b1 = *(const bf16x16*)(brow1 + k + g * 16);
    bf16x16 b2 = *(const bf16x16*)(brow2 + k + g * 16);
    bf16x16 b3 = *(const bf16x16*)(brow3 + k + g * 16);
    acc[0][0] = wmma_bf16(a0, b0, acc[0][0]);
    acc[0][1] = wmma_bf16(a0, b1, acc[0][1]);
    acc[0][2] = wmma_bf16(a0, b2, acc[0][2]);
    acc[0][3] = wmma_bf16(a0, b3, acc[0][3]);
    acc[1][0] = wmma_bf16(a1, b0, acc[1][0]);
    acc[1][1] = wmma_bf16(a1, b1, acc[1][1]);
    acc[1][2] = wmma_bf16(a1, b2, acc[1][2]);
    acc[1][3] = wmma_bf16(a1, b3, acc[1][3]);
  }

  // ---- RoPE epilogue: d and d+32 are acc[.][s] and acc[.][s+2], same lane,
  //      same accumulator index -> pure register math.
  const int which = nblk / 16;   // 0=q 1=k 2=v
  const int h     = nblk % 16;

#pragma unroll
  for (int ma = 0; ma < 2; ++ma) {
#pragma unroll
    for (int i = 0; i < 8; ++i) {
      const int gm = m0 + ma * 16 + i + 8 * g;   // global row
      const int b  = gm >> 11;                   // /2048
      const int t  = gm & 2047;
      const size_t bh = (size_t)(b * Hh + h);
#pragma unroll
      for (int s = 0; s < 2; ++s) {
        const int   dlo = s * 16 + ln;           // < 32
        const int   dhi = dlo + 32;
        const float flo = freqs[t * HD + dlo];
        const float fhi = freqs[t * HD + dhi];
        const float x1  = acc[ma][s][i];         // x[..., dlo]
        const float x2  = acc[ma][s + 2][i];     // x[..., dhi]
        const __bf16 vlo = (__bf16)(x1 * __cosf(flo) - x2 * __sinf(flo));
        const __bf16 vhi = (__bf16)(x2 * __cosf(fhi) + x1 * __sinf(fhi));
        if (which == 0) {
          qout[(bh * Tt + t) * HD + dlo] = vlo;
          qout[(bh * Tt + t) * HD + dhi] = vhi;
        } else if (which == 1) {
          kout[(bh * Tt + t) * HD + dlo] = vlo;
          kout[(bh * Tt + t) * HD + dhi] = vhi;
        } else {
          vtout[(bh * HD + dlo) * Tt + t] = vlo;
          vtout[(bh * HD + dhi) * Tt + t] = vhi;
        }
      }
    }
  }
}

// ---------------------------------------------------------------------------
// Kernel 2: flash attention. One wave per 16-row q-tile of one (b,h).
//   block = 256 threads (8 waves), grid = B*H*(T/16)/8 = 512
//   per 32-key step: 4 wmma for S = Q K^T, online softmax, 4 wmma for P V.
//   P transposed C-layout -> A-layout through per-wave LDS (same-wave LDS
//   ops are in-order on CDNA5, no barrier needed).
// ---------------------------------------------------------------------------
__global__ __launch_bounds__(256)
void flash_attn_kernel(const __bf16* __restrict__ q,
                       const __bf16* __restrict__ k,
                       const __bf16* __restrict__ vT,
                       const unsigned char* __restrict__ mask,
                       __bf16* __restrict__ yb) {
  __shared__ float lds_p[8][16][33];   // [wave][q-row][key 0..31], padded

  const int lane = threadIdx.x & 31;
  const int wv   = threadIdx.x >> 5;
  const int g    = lane >> 4;
  const int ln   = lane & 15;

  const int gw = blockIdx.x * 8 + wv;  // 0..4095
  const int qt = gw & 127;             // q-tile
  const int bh = gw >> 7;              // 0..31
  const int b  = bh >> 4;
  const int h  = bh & 15;
  const int t0 = qt * 16;

  const __bf16* qb = q  + (size_t)bh * Tt * HD;
  const __bf16* kb = k  + (size_t)bh * Tt * HD;
  const __bf16* vb = vT + (size_t)bh * HD * Tt;
  const unsigned char* mrow = mask + (size_t)b * Tt;

  // Q A-fragments for d in [0,32) and [32,64)
  const __bf16* qrow = qb + (size_t)(t0 + ln) * HD;
  const int c0 = g * 8;
  const bf16x16 A0 = ld_afrag(qrow, c0);
  const bf16x16 A1 = ld_afrag(qrow + 32, c0);

  f32x8 o0 = {}, o1 = {}, o2 = {}, o3 = {};
  float rmax[8], rsum[8];
#pragma unroll
  for (int i = 0; i < 8; ++i) { rmax[i] = -3.0e38f; rsum[i] = 0.0f; }

  for (int kt = 0; kt < Tt; kt += 32) {
    // ---- S = Q K^T for 32 keys (two 16x16 tiles) ----
    f32x8 s0 = {}, s1 = {};
    {
      const __bf16* kr0 = kb + (size_t)(kt + ln) * HD + g * 16;
      s0 = wmma_bf16(A0, *(const bf16x16*)kr0, s0);
      s0 = wmma_bf16(A1, *(const bf16x16*)(kr0 + 32), s0);
      const __bf16* kr1 = kb + (size_t)(kt + 16 + ln) * HD + g * 16;
      s1 = wmma_bf16(A0, *(const bf16x16*)kr1, s1);
      s1 = wmma_bf16(A1, *(const bf16x16*)(kr1 + 32), s1);
    }

    const bool msk0 = mrow[kt + ln] != 0;        // key-axis mask
    const bool msk1 = mrow[kt + 16 + ln] != 0;

    // ---- online softmax, row r = i + 8g lives in this lane's group ----
#pragma unroll
    for (int i = 0; i < 8; ++i) {
      float a = msk0 ? NEG_BIG_F : s0[i] * SCALE_F;
      float c = msk1 ? NEG_BIG_F : s1[i] * SCALE_F;
      float tmax = fmaxf(a, c);
      tmax = fmaxf(tmax, __shfl_xor(tmax, 1, 16));
      tmax = fmaxf(tmax, __shfl_xor(tmax, 2, 16));
      tmax = fmaxf(tmax, __shfl_xor(tmax, 4, 16));
      tmax = fmaxf(tmax, __shfl_xor(tmax, 8, 16));
      const float nm    = fmaxf(rmax[i], tmax);
      const float alpha = __expf(rmax[i] - nm);
      rmax[i] = nm;
      const float p0 = __expf(a - nm);
      const float p1 = __expf(c - nm);
      float ps = p0 + p1;
      ps += __shfl_xor(ps, 1, 16);
      ps += __shfl_xor(ps, 2, 16);
      ps += __shfl_xor(ps, 4, 16);
      ps += __shfl_xor(ps, 8, 16);
      rsum[i] = rsum[i] * alpha + ps;
      o0[i] *= alpha; o1[i] *= alpha; o2[i] *= alpha; o3[i] *= alpha;
      const int r = i + 8 * g;
      lds_p[wv][r][ln]      = p0;
      lds_p[wv][r][16 + ln] = p1;
    }

    // ---- P (C layout) -> A layout via LDS; same-wave LDS is in-order ----
    bf16x16 pa;
#pragma unroll
    for (int e = 0; e < 8; ++e) {
      pa[e]     = (__bf16)lds_p[wv][ln][2 * c0 + e];
      pa[8 + e] = (__bf16)lds_p[wv][ln][2 * c0 + 8 + e];
    }

    // ---- O += P V  (4 dim-slices of 16) ----
    {
      const __bf16* vr0 = vb + (size_t)(0 * 16 + ln) * Tt + kt + g * 16;
      const __bf16* vr1 = vb + (size_t)(1 * 16 + ln) * Tt + kt + g * 16;
      const __bf16* vr2 = vb + (size_t)(2 * 16 + ln) * Tt + kt + g * 16;
      const __bf16* vr3 = vb + (size_t)(3 * 16 + ln) * Tt + kt + g * 16;
      o0 = wmma_bf16(pa, *(const bf16x16*)vr0, o0);
      o1 = wmma_bf16(pa, *(const bf16x16*)vr1, o1);
      o2 = wmma_bf16(pa, *(const bf16x16*)vr2, o2);
      o3 = wmma_bf16(pa, *(const bf16x16*)vr3, o3);
    }
  }

  // ---- normalize and store y (b, t, h*64+d) as bf16 ----
#pragma unroll
  for (int i = 0; i < 8; ++i) {
    const float inv = 1.0f / rsum[i];
    const int   t   = t0 + i + 8 * g;
    const size_t row = ((size_t)b * Tt + t) * Cc + h * HD;
    yb[row + 0 * 16 + ln] = (__bf16)(o0[i] * inv);
    yb[row + 1 * 16 + ln] = (__bf16)(o1[i] * inv);
    yb[row + 2 * 16 + ln] = (__bf16)(o2[i] * inv);
    yb[row + 3 * 16 + ln] = (__bf16)(o3[i] * inv);
  }
}

// ---------------------------------------------------------------------------
// Kernel 3: output projection, 32x64 tile per wave.
//   out[m,n] = sum_k y[m,k] W_proj[n,k] + b[n]
//   waves = (4096/32) * (1024/64) = 2048; block = 256 (8 waves) -> 256 blocks
// ---------------------------------------------------------------------------
__global__ __launch_bounds__(256)
void proj_kernel(const __bf16* __restrict__ yb,
                 const __bf16* __restrict__ wpb,
                 const float*  __restrict__ bproj,
                 float* __restrict__ out) {
  const int lane = threadIdx.x & 31;
  const int wv   = threadIdx.x >> 5;
  const int g    = lane >> 4;
  const int ln   = lane & 15;

  const int gw   = blockIdx.x * 8 + wv;   // 0..2047
  const int nblk = gw & 15;
  const int mblk = gw >> 4;
  const int m0   = mblk * 32;
  const int n0   = nblk * 64;

  const __bf16* arow0 = yb + (size_t)(m0 + ln) * Cc;
  const __bf16* arow1 = yb + (size_t)(m0 + 16 + ln) * Cc;
  const __bf16* brow0 = wpb + (size_t)(n0 +  0 + ln) * Cc;
  const __bf16* brow1 = wpb + (size_t)(n0 + 16 + ln) * Cc;
  const __bf16* brow2 = wpb + (size_t)(n0 + 32 + ln) * Cc;
  const __bf16* brow3 = wpb + (size_t)(n0 + 48 + ln) * Cc;
  const int c0 = g * 8;

  f32x8 acc[2][4];
#pragma unroll
  for (int ma = 0; ma < 2; ++ma)
#pragma unroll
    for (int s = 0; s < 4; ++s) acc[ma][s] = (f32x8){};

  for (int k = 0; k < Cc; k += 32) {
    __builtin_prefetch(arow0 + k + 128, 0, 3);
    __builtin_prefetch(brow0 + k + 128, 0, 3);
    bf16x16 a0 = ld_afrag(arow0 + k, c0);
    bf16x16 a1 = ld_afrag(arow1 + k, c0);
    bf16x16 b0 = *(const bf16x16*)(brow0 + k + g * 16);
    bf16x16 b1 = *(const bf16x16*)(brow1 + k + g * 16);
    bf16x16 b2 = *(const bf16x16*)(brow2 + k + g * 16);
    bf16x16 b3 = *(const bf16x16*)(brow3 + k + g * 16);
    acc[0][0] = wmma_bf16(a0, b0, acc[0][0]);
    acc[0][1] = wmma_bf16(a0, b1, acc[0][1]);
    acc[0][2] = wmma_bf16(a0, b2, acc[0][2]);
    acc[0][3] = wmma_bf16(a0, b3, acc[0][3]);
    acc[1][0] = wmma_bf16(a1, b0, acc[1][0]);
    acc[1][1] = wmma_bf16(a1, b1, acc[1][1]);
    acc[1][2] = wmma_bf16(a1, b2, acc[1][2]);
    acc[1][3] = wmma_bf16(a1, b3, acc[1][3]);
  }

#pragma unroll
  for (int s = 0; s < 4; ++s) {
    const float bn = bproj[n0 + s * 16 + ln];
#pragma unroll
    for (int ma = 0; ma < 2; ++ma) {
#pragma unroll
      for (int i = 0; i < 8; ++i) {
        const int gm = m0 + ma * 16 + i + 8 * g;
        out[(size_t)gm * Cc + n0 + s * 16 + ln] = acc[ma][s][i] + bn;
      }
    }
  }
}

// ---------------------------------------------------------------------------
// Host launcher
// ---------------------------------------------------------------------------
extern "C" void kernel_launch(void* const* d_in, const int* in_sizes, int n_in,
                              void* d_out, int out_size, void* d_ws, size_t ws_size,
                              hipStream_t stream) {
  (void)in_sizes; (void)n_in; (void)out_size; (void)ws_size;

  const float*         x     = (const float*)d_in[0];         // (2,2048,1024)
  const float*         freqs = (const float*)d_in[1];         // (1,1,2048,64)
  const unsigned char* mask  = (const unsigned char*)d_in[2]; // (2,2048) bool
  const float*         Wa    = (const float*)d_in[3];         // (3072,1024)
  const float*         Wp    = (const float*)d_in[4];         // (1024,1024)
  const float*         bp    = (const float*)d_in[5];         // (1024,)
  float*               out   = (float*)d_out;                 // (2,2048,1024)

  // workspace carve-up (all region sizes are multiples of 256B)
  char* p = (char*)d_ws;
  __bf16* xb  = (__bf16*)p; p += (size_t)Bb * Tt * Cc * 2;       // 8 MB
  __bf16* wab = (__bf16*)p; p += (size_t)3 * Cc * Cc * 2;        // 6 MB
  __bf16* wpb = (__bf16*)p; p += (size_t)Cc * Cc * 2;            // 2 MB
  __bf16* qb  = (__bf16*)p; p += (size_t)Bb * Hh * Tt * HD * 2;  // 8 MB
  __bf16* kb  = (__bf16*)p; p += (size_t)Bb * Hh * Tt * HD * 2;  // 8 MB
  __bf16* vtb = (__bf16*)p; p += (size_t)Bb * Hh * HD * Tt * 2;  // 8 MB
  __bf16* yb  = (__bf16*)p;                                      // 8 MB

  const int nX  = Bb * Tt * Cc;      // 4194304
  const int nWa = 3 * Cc * Cc;       // 3145728
  const int nWp = Cc * Cc;           // 1048576

  cvt_bf16_kernel<<<(nX  + 255) / 256, 256, 0, stream>>>(x,  xb,  nX);
  cvt_bf16_kernel<<<(nWa + 255) / 256, 256, 0, stream>>>(Wa, wab, nWa);
  cvt_bf16_kernel<<<(nWp + 255) / 256, 256, 0, stream>>>(Wp, wpb, nWp);

  // QKV GEMM + RoPE: (4096/32)*(3072/64) = 6144 waves / 8 per block
  qkv_rope_kernel<<<768, 256, 0, stream>>>(xb, wab, freqs, qb, kb, vtb);

  // Flash attention: 2*16*(2048/16) = 4096 waves / 8 per block
  flash_attn_kernel<<<512, 256, 0, stream>>>(qb, kb, vtb, mask, yb);

  // Projection + bias: (4096/32)*(1024/64) = 2048 waves / 8 per block
  proj_kernel<<<256, 256, 0, stream>>>(yb, wpb, bp, out);
}
